// MassConservingLSTMCell_80839874445758
// MI455X (gfx1250) — matrix operand order repaired and checked by reading, not compiled, not run on hardware
//
#include <hip/hip_runtime.h>
#include <hip/hip_bf16.h>
#include <math.h>

typedef __attribute__((ext_vector_type(16))) _Float16 v16h;
typedef __attribute__((ext_vector_type(8)))  _Float16 v8h;
typedef __attribute__((ext_vector_type(8)))  float    v8f;

#define B_DIM 1024
#define U_DIM 256
#define F_DIM 31
#define D_DIM 288   // 1 + 256 + 31, exactly 9*32

// workspace layout (bytes), all 16B aligned
#define OFF_FEAT 0ull                               // 1024*288*2   = 589824
#define OFF_WRT  589824ull                          // 65536*288*2  = 37748736
#define OFF_WIT  (OFF_WRT + 37748736ull)            // 256*288*2    = 147456
#define OFF_WOT  (OFF_WIT + 147456ull)              // 256*288*2    = 147456
#define OFF_CS   (OFF_WOT + 147456ull)              // 1024*256*4   = 1048576

union FragAB { v16h v; v8h h2[2]; };

// ---------------------------------------------------------------------------
// Convert f32 weight [K][N] (row-major) into f16 transposed [N][K] layout.
// ---------------------------------------------------------------------------
__global__ void convert_transpose_kernel(const float* __restrict__ src,
                                         _Float16* __restrict__ dst,
                                         int K, int N) {
  int n = blockIdx.x * blockDim.x + threadIdx.x;
  if (n >= N) return;
  for (int k = 0; k < K; k += 8) {
    v8h buf;
#pragma unroll
    for (int j = 0; j < 8; ++j)
      buf[j] = (_Float16)src[(size_t)(k + j) * N + n];
    *(v8h*)&dst[(size_t)n * K + k] = buf;
  }
}

// ---------------------------------------------------------------------------
// Prep: l1-normalize last_cell, assemble f16 feature rows, zero cell_sys acc.
// grid = 1024 (one block per batch row), block = 256
// ---------------------------------------------------------------------------
__global__ void prep_kernel(const float* __restrict__ last_cell,
                            const float* __restrict__ precip,
                            const float* __restrict__ other,
                            _Float16* __restrict__ feat,
                            float* __restrict__ cell_sys) {
  __shared__ float red[256];
  const int b = blockIdx.x;
  const int t = threadIdx.x;
  const float lc = last_cell[b * U_DIM + t];
  red[t] = fabsf(lc);
  __syncthreads();
  for (int s = 128; s > 0; s >>= 1) {
    if (t < s) red[t] += red[t + s];
    __syncthreads();
  }
  const float inv_l1 = 1.0f / (red[0] + 1e-7f);

  _Float16* row = feat + (size_t)b * D_DIM;
  if (t == 0) row[0] = (_Float16)precip[b];
  row[1 + t] = (_Float16)(lc * inv_l1);
  if (t < F_DIM) row[1 + U_DIM + t] = (_Float16)other[b * F_DIM + t];
  cell_sys[b * U_DIM + t] = 0.0f;
}

// ---------------------------------------------------------------------------
// Fused R-GEMM + relu + row-L1-normalize + einsum('bu,buv->bv') partials.
// grid = (16 u-chunks, 64 batch tiles), block = 256 (8 waves).
// Wave w owns v-tiles {2w, 2w+1}; per u: 2 tiles x 9 WMMA (K=288).
// A-fragments (features tile) hoisted into registers across the u-loop;
// __launch_bounds__(256, 1) raises the VGPR ceiling so they don't spill.
// ---------------------------------------------------------------------------
__global__ void __launch_bounds__(256, 1)
einsum_kernel(const _Float16* __restrict__ feat,
              const _Float16* __restrict__ wrt,   // [65536][288]
              const float* __restrict__ b_r,
              const float* __restrict__ last_cell,
              float* __restrict__ cell_sys) {
  __shared__ _Float16 sA[16 * D_DIM];
  __shared__ float sLC[16 * 16];
  __shared__ float sRow[16];
  __shared__ float sScale[16];

  const int u0 = blockIdx.x * 16;
  const int b0 = blockIdx.y * 16;
  const int t  = threadIdx.x;

  for (int i = t; i < 16 * D_DIM; i += 256) {
    int m = i / D_DIM, k = i % D_DIM;
    sA[i] = feat[(size_t)(b0 + m) * D_DIM + k];
  }
  {
    int m = t / 16, j = t % 16;
    sLC[t] = last_cell[(b0 + m) * U_DIM + u0 + j];
  }
  if (t < 16) sRow[t] = 0.0f;
  __syncthreads();

  const int lane = t & 31;
  const int wave = t >> 5;
  const int half = lane >> 4;
  const int l16  = lane & 15;
  const int vb0  = wave * 32;
  const int vb1  = vb0 + 16;

  // Hoist the 9 A-fragments (invariant across the entire u-loop).
  FragAB afr[9];
#pragma unroll
  for (int kk = 0; kk < 9; ++kk) {
    const int abase = l16 * D_DIM + kk * 32 + half * 16;
    afr[kk].h2[0] = *(const v8h*)&sA[abase];
    afr[kk].h2[1] = *(const v8h*)&sA[abase + 8];
  }

  // Incremental W_r column pointers: advance by U_DIM rows of K each u-step.
  const _Float16* p0 = wrt + (size_t)(u0 * U_DIM + vb0 + l16) * D_DIM;
  const _Float16* p1 = wrt + (size_t)(u0 * U_DIM + vb1 + l16) * D_DIM;
  const size_t ustep = (size_t)U_DIM * D_DIM;

  v8f acc0 = {0.f,0.f,0.f,0.f,0.f,0.f,0.f,0.f};
  v8f acc1 = {0.f,0.f,0.f,0.f,0.f,0.f,0.f,0.f};

  for (int j = 0; j < 16; ++j) {
    const int u = u0 + j;

    v8f c0 = {0.f,0.f,0.f,0.f,0.f,0.f,0.f,0.f};
    v8f c1 = {0.f,0.f,0.f,0.f,0.f,0.f,0.f,0.f};
#pragma unroll
    for (int kk = 0; kk < 9; ++kk) {
      const int koff = kk * 32 + half * 16;
      FragAB bf0, bf1;
      bf0.h2[0] = *(const v8h*)&p0[koff];
      bf0.h2[1] = *(const v8h*)&p0[koff + 8];
      bf1.h2[0] = *(const v8h*)&p1[koff];
      bf1.h2[1] = *(const v8h*)&p1[koff + 8];
      c0 = __builtin_amdgcn_wmma_f32_16x16x32_f16(false, afr[kk].v, false, bf0.v,
                                                  (short)0, c0, false, false);
      c1 = __builtin_amdgcn_wmma_f32_16x16x32_f16(false, afr[kk].v, false, bf1.v,
                                                  (short)0, c1, false, false);
    }
    p0 += ustep;
    p1 += ustep;

    const float bb0 = b_r[u * U_DIM + vb0 + l16];
    const float bb1 = b_r[u * U_DIM + vb1 + l16];
    float rs[8];
#pragma unroll
    for (int r = 0; r < 8; ++r) {
      float x0 = fmaxf(c0[r] + bb0, 0.0f);
      float x1 = fmaxf(c1[r] + bb1, 0.0f);
      c0[r] = x0; c1[r] = x1;
      rs[r] = x0 + x1;
    }
    // reduce across the 16 lanes of each half (rows M = r + 8*half)
#pragma unroll
    for (int r = 0; r < 8; ++r) {
#pragma unroll
      for (int off = 1; off < 16; off <<= 1)
        rs[r] += __shfl_xor(rs[r], off, 16);
    }
#pragma unroll
    for (int r = 0; r < 8; ++r)
      if (l16 == r) atomicAdd(&sRow[r + 8 * half], rs[r]);
    __syncthreads();
    if (t < 16) {
      sScale[t] = sLC[t * 16 + j] / (sRow[t] + 1e-30f);
      sRow[t] = 0.0f;
    }
    __syncthreads();
#pragma unroll
    for (int r = 0; r < 8; ++r) {
      const float sc = sScale[r + 8 * half];
      acc0[r] += sc * c0[r];
      acc1[r] += sc * c1[r];
    }
  }

#pragma unroll
  for (int r = 0; r < 8; ++r) {
    const int brow = b0 + r + 8 * half;
    atomicAdd(&cell_sys[brow * U_DIM + vb0 + l16], acc0[r]);
    atomicAdd(&cell_sys[brow * U_DIM + vb1 + l16], acc1[r]);
  }
}

// ---------------------------------------------------------------------------
// Gates (WMMA) + final combine.  grid = 64 batch tiles, block = 256 (8 waves).
// ---------------------------------------------------------------------------
__global__ void __launch_bounds__(256, 1)
gates_kernel(const _Float16* __restrict__ feat,
             const _Float16* __restrict__ wit,  // [256][288]
             const _Float16* __restrict__ wot,  // [256][288]
             const float* __restrict__ b_i,
             const float* __restrict__ b_o,
             const float* __restrict__ precip,
             const float* __restrict__ cell_sys,
             float* __restrict__ out) {
  __shared__ _Float16 sA[16 * D_DIM];
  __shared__ float sSum[16];
  __shared__ float sP[16];

  const int b0 = blockIdx.x * 16;
  const int t  = threadIdx.x;

  for (int i = t; i < 16 * D_DIM; i += 256) {
    int m = i / D_DIM, k = i % D_DIM;
    sA[i] = feat[(size_t)(b0 + m) * D_DIM + k];
  }
  if (t < 16) { sSum[t] = 0.0f; sP[t] = precip[b0 + t]; }
  __syncthreads();

  const int lane = t & 31;
  const int wave = t >> 5;
  const int half = lane >> 4;
  const int l16  = lane & 15;
  const int vb0  = wave * 32;
  const int vb1  = vb0 + 16;
  const size_t ci0 = (size_t)(vb0 + l16) * D_DIM;
  const size_t ci1 = (size_t)(vb1 + l16) * D_DIM;

  v8f gi0 = {0.f,0.f,0.f,0.f,0.f,0.f,0.f,0.f};
  v8f gi1 = {0.f,0.f,0.f,0.f,0.f,0.f,0.f,0.f};
  v8f go0 = {0.f,0.f,0.f,0.f,0.f,0.f,0.f,0.f};
  v8f go1 = {0.f,0.f,0.f,0.f,0.f,0.f,0.f,0.f};
#pragma unroll
  for (int kk = 0; kk < 9; ++kk) {
    const int k0 = kk * 32;
    FragAB a;
    const int abase = l16 * D_DIM + k0 + half * 16;
    a.h2[0] = *(const v8h*)&sA[abase];
    a.h2[1] = *(const v8h*)&sA[abase + 8];
    const int koff = k0 + half * 16;
    FragAB wi0, wi1, wo0, wo1;
    wi0.h2[0] = *(const v8h*)&wit[ci0 + koff];
    wi0.h2[1] = *(const v8h*)&wit[ci0 + koff + 8];
    wi1.h2[0] = *(const v8h*)&wit[ci1 + koff];
    wi1.h2[1] = *(const v8h*)&wit[ci1 + koff + 8];
    wo0.h2[0] = *(const v8h*)&wot[ci0 + koff];
    wo0.h2[1] = *(const v8h*)&wot[ci0 + koff + 8];
    wo1.h2[0] = *(const v8h*)&wot[ci1 + koff];
    wo1.h2[1] = *(const v8h*)&wot[ci1 + koff + 8];
    gi0 = __builtin_amdgcn_wmma_f32_16x16x32_f16(false, a.v, false, wi0.v,
                                                 (short)0, gi0, false, false);
    gi1 = __builtin_amdgcn_wmma_f32_16x16x32_f16(false, a.v, false, wi1.v,
                                                 (short)0, gi1, false, false);
    go0 = __builtin_amdgcn_wmma_f32_16x16x32_f16(false, a.v, false, wo0.v,
                                                 (short)0, go0, false, false);
    go1 = __builtin_amdgcn_wmma_f32_16x16x32_f16(false, a.v, false, wo1.v,
                                                 (short)0, go1, false, false);
  }

  const float bi0 = b_i[vb0 + l16], bi1 = b_i[vb1 + l16];
  const float bo0 = b_o[vb0 + l16], bo1 = b_o[vb1 + l16];
  float rs[8];
#pragma unroll
  for (int r = 0; r < 8; ++r) {
    gi0[r] = 1.0f / (1.0f + __expf(-(gi0[r] + bi0)));
    gi1[r] = 1.0f / (1.0f + __expf(-(gi1[r] + bi1)));
    go0[r] = 1.0f / (1.0f + __expf(-(go0[r] + bo0)));
    go1[r] = 1.0f / (1.0f + __expf(-(go1[r] + bo1)));
    rs[r] = gi0[r] + gi1[r];
  }
#pragma unroll
  for (int r = 0; r < 8; ++r) {
#pragma unroll
    for (int off = 1; off < 16; off <<= 1)
      rs[r] += __shfl_xor(rs[r], off, 16);
  }
#pragma unroll
  for (int r = 0; r < 8; ++r)
    if (l16 == r) atomicAdd(&sSum[r + 8 * half], rs[r]);
  __syncthreads();

  const size_t BU = (size_t)B_DIM * U_DIM;
#pragma unroll
  for (int r = 0; r < 8; ++r) {
    const int M = r + 8 * half;
    const int b = b0 + M;
    const float inv_s = 1.0f / sSum[M];
    const float pb = sP[M];

    float cs0 = cell_sys[b * U_DIM + vb0 + l16];
    float cand0 = gi0[r] * inv_s * pb + cs0;
    out[(size_t)b * U_DIM + vb0 + l16]      = (1.0f - go0[r]) * cand0; // cell
    out[BU + (size_t)b * U_DIM + vb0 + l16] = go0[r] * cand0;          // output

    float cs1 = cell_sys[b * U_DIM + vb1 + l16];
    float cand1 = gi1[r] * inv_s * pb + cs1;
    out[(size_t)b * U_DIM + vb1 + l16]      = (1.0f - go1[r]) * cand1;
    out[BU + (size_t)b * U_DIM + vb1 + l16] = go1[r] * cand1;
  }
}

// ---------------------------------------------------------------------------
extern "C" void kernel_launch(void* const* d_in, const int* in_sizes, int n_in,
                              void* d_out, int out_size, void* d_ws, size_t ws_size,
                              hipStream_t stream) {
  (void)in_sizes; (void)n_in; (void)out_size; (void)ws_size;
  const float* last_cell = (const float*)d_in[0];
  const float* precip    = (const float*)d_in[1];
  const float* other     = (const float*)d_in[2];
  const float* W_i       = (const float*)d_in[3];
  const float* b_i       = (const float*)d_in[4];
  const float* W_r       = (const float*)d_in[5];
  const float* b_r       = (const float*)d_in[6];
  const float* W_o       = (const float*)d_in[7];
  const float* b_o       = (const float*)d_in[8];
  float* out = (float*)d_out;

  char* ws = (char*)d_ws;
  _Float16* feat = (_Float16*)(ws + OFF_FEAT);
  _Float16* wrt  = (_Float16*)(ws + OFF_WRT);
  _Float16* wit  = (_Float16*)(ws + OFF_WIT);
  _Float16* wot  = (_Float16*)(ws + OFF_WOT);
  float* cell_sys = (float*)(ws + OFF_CS);

  convert_transpose_kernel<<<256, 256, 0, stream>>>(W_r, wrt, D_DIM, U_DIM * U_DIM);
  convert_transpose_kernel<<<1, 256, 0, stream>>>(W_i, wit, D_DIM, U_DIM);
  convert_transpose_kernel<<<1, 256, 0, stream>>>(W_o, wot, D_DIM, U_DIM);
  prep_kernel<<<B_DIM, 256, 0, stream>>>(last_cell, precip, other, feat, cell_sys);
  einsum_kernel<<<dim3(16, 64), 256, 0, stream>>>(feat, wrt, b_r, last_cell, cell_sys);
  gates_kernel<<<64, 256, 0, stream>>>(feat, wit, wot, b_i, b_o, precip, cell_sys, out);
}
